// CrossAttention_20504173871326
// MI455X (gfx1250) — matrix operand order repaired and checked by reading, not compiled, not run on hardware
//
#include <hip/hip_runtime.h>
#include <hip/hip_bf16.h>

// ---------------------------------------------------------------------------
// CDNA5 (gfx1250, wave32) fused cross-attention:
//   1) qkv_proj_kernel : X[S*B,100] x W[100,128] -> f16 [B,S,128]   (WMMA)
//   2) flash_attn_kernel: online-softmax flash attention             (WMMA)
//      with double-buffered LDS tiles and CDNA5 async global->LDS
//      prefetch of the K tile (inline asm, tracked by ASYNCcnt).
// Fragment construction follows CDNA5 ISA layouts:
//   A (16x32 f16): lane L -> row L&15; slots 0-7 = K 8h..8h+7,
//                  slots 8-15 = K 16+8h..16+8h+7  (h = L>>4)  [contiguous]
//   B (32x16 f16): lane L -> col L&15; slot j<8 = K 8h+j, j>=8 = K 8+8h+j        [contiguous]
//   C/D (16x16 f32): lane L -> col L&15; elem j = row j+8h
// ---------------------------------------------------------------------------

typedef __attribute__((ext_vector_type(16))) _Float16 v16h;
typedef __attribute__((ext_vector_type(8)))  _Float16 v8h;
typedef __attribute__((ext_vector_type(8)))  float    v8f;

union V16H { v16h v; v8h h[2]; };

#define S1N   2048
#define S2N   2048
#define BATCH 16
#define DHN   100
#define DKN   128
#define KTILE 32

#if defined(__has_builtin)
#if __has_builtin(__builtin_amdgcn_s_wait_asynccnt)
#define USE_ASYNC_COPY 1
#endif
#endif

#ifdef USE_ASYNC_COPY
// 16B global -> LDS async copy (per-lane), tracked by ASYNCcnt.
// Generic LDS pointers carry the LDS byte offset in their low 32 bits
// (ISA 10.2 aperture rules), which is exactly the VDST operand.
__device__ __forceinline__ void async_ld16(const _Float16* g, _Float16* l) {
  unsigned lds_off = (unsigned)(size_t)l;
  unsigned long long ga = (unsigned long long)g;
  asm volatile("global_load_async_to_lds_b128 %0, %1, off"
               :: "v"(lds_off), "v"(ga)
               : "memory");
}
__device__ __forceinline__ void wait_async0() {
  __builtin_amdgcn_s_wait_asynccnt(0);
}
#endif

__device__ __forceinline__ v8f wmma_f16(v16h a, v16h b, v8f c) {
  return __builtin_amdgcn_wmma_f32_16x16x32_f16(
      /*neg_a=*/false, a, /*neg_b=*/false, b,
      /*c_mod=*/(short)0, c, /*reuse_a=*/false, /*reuse_b=*/false);
}

// ---------------------------------------------------------------------------
// Projection: Y[b*S + s][dk] = sum_d X[s*B+b][d] * W[d][dk]  (f16 out)
// Block = 128 threads = 4 waves; each wave does a 16-row tile, 128 cols.
// ---------------------------------------------------------------------------
__global__ __launch_bounds__(128) void qkv_proj_kernel(
    const float* __restrict__ X, const float* __restrict__ W,
    _Float16* __restrict__ Y, int S)
{
  // W transposed to [col][k], k zero-padded to 128, row stride 136 (bank skew)
  __shared__ __align__(16) _Float16 Wt[DKN * 136];

  const int tid  = threadIdx.x;
  const int lane = tid & 31;
  const int wave = tid >> 5;
  const int h    = lane >> 4;
  const int col  = lane & 15;

  {
    const int c = tid;                      // 128 threads <-> 128 cols
    for (int k = 0; k < DKN; ++k) {
      float w = (k < DHN) ? W[k * DKN + c] : 0.0f;
      Wt[c * 136 + k] = (_Float16)w;
    }
  }
  __syncthreads();

  const int rbase = (blockIdx.x * 4 + wave) * 16;   // row tile in [0, S*B)
  const int r     = rbase + col;                    // this lane's A-row
  const float* xr = X + (size_t)r * DHN;

  // A fragments over 4 k-chunks of 32 (DH=100 zero-padded)
  v16h afrag[4];
  #pragma unroll
  for (int kc = 0; kc < 4; ++kc) {
    V16H a;
    const int k0 = kc * 32 + 8 * h;
    #pragma unroll
    for (int i = 0; i < 8; ++i) {
      const int ka = k0 + i, kb = k0 + 16 + i;
      a.v[i]     = (ka < DHN) ? (_Float16)xr[ka] : (_Float16)0.0f;
      a.v[8 + i] = (kb < DHN) ? (_Float16)xr[kb] : (_Float16)0.0f;
    }
    afrag[kc] = a.v;
  }

  v8f zero = {};
  v8f acc[8];
  #pragma unroll
  for (int cc = 0; cc < 8; ++cc) acc[cc] = zero;

  #pragma unroll
  for (int kc = 0; kc < 4; ++kc) {
    // batch-load all 8 B fragments, then run the WMMA chain
    v16h bf[8];
    #pragma unroll
    for (int cc = 0; cc < 8; ++cc) {
      V16H bfr;
      const _Float16* bp = &Wt[(cc * 16 + col) * 136 + kc * 32 + 8 * h];
      bfr.h[0] = *(const v8h*)bp;
      bfr.h[1] = *(const v8h*)(bp + 16);
      bf[cc] = bfr.v;
    }
    #pragma unroll
    for (int cc = 0; cc < 8; ++cc)
      acc[cc] = wmma_f16(afrag[kc], bf[cc], acc[cc]);
  }

  // C layout: elem j of lane L -> row rbase + j + 8h, col cc*16 + (L&15)
  #pragma unroll
  for (int cc = 0; cc < 8; ++cc) {
    #pragma unroll
    for (int j = 0; j < 8; ++j) {
      const int ro = rbase + j + 8 * h;     // X row index = s*BATCH + b
      const int s  = ro >> 4;
      const int b  = ro & 15;
      Y[((size_t)b * S + s) * DKN + cc * 16 + col] = (_Float16)acc[cc][j];
    }
  }
}

// ---------------------------------------------------------------------------
// Flash attention. Block = 128 threads = 4 waves; block owns (batch b,
// 64 Q rows); wave owns 16 Q rows. Loop over S2 in 32-key tiles with
// double-buffered LDS and pipelined prefetch of the next tile.
// ---------------------------------------------------------------------------
__global__ __launch_bounds__(128) void flash_attn_kernel(
    const _Float16* __restrict__ Q, const _Float16* __restrict__ K,
    const _Float16* __restrict__ V, float* __restrict__ Out)
{
  __shared__ __align__(16) _Float16 Kt[2][KTILE * 136]; // [key][dim] row-major
  __shared__ __align__(16) _Float16 Vt[2][DKN * 40];    // [dim][key] transposed
  __shared__ __align__(16) _Float16 Pl[4 * 16 * 40];    // per-wave P tiles

  const int tid  = threadIdx.x;
  const int lane = tid & 31;
  const int wave = tid >> 5;
  const int h    = lane >> 4;
  const int col  = lane & 15;
  const int b    = blockIdx.y;
  const int qbase = blockIdx.x * 64 + wave * 16;

  const _Float16* Kg = K + (size_t)b * S2N * DKN;
  const _Float16* Vg = V + (size_t)b * S2N * DKN;

  v8h kstage[4], vstage[4];

  // ---- prologue: tile 0 -> buffer 0 ----
  #pragma unroll
  for (int i = 0; i < 4; ++i) {
    const int chunk = tid * 4 + i;
    const int r = chunk >> 4, c = chunk & 15;
#ifdef USE_ASYNC_COPY
    async_ld16(Kg + (size_t)r * DKN + c * 8, &Kt[0][r * 136 + c * 8]);
#else
    kstage[i] = *(const v8h*)(Kg + (size_t)r * DKN + c * 8);
#endif
    vstage[i] = *(const v8h*)(Vg + (size_t)r * DKN + c * 8);
  }
  #pragma unroll
  for (int i = 0; i < 4; ++i) {
    const int chunk = tid * 4 + i;
    const int r = chunk >> 4, c = chunk & 15;
#ifndef USE_ASYNC_COPY
    *(v8h*)&Kt[0][r * 136 + c * 8] = kstage[i];
#endif
    #pragma unroll
    for (int e = 0; e < 8; ++e)
      Vt[0][(c * 8 + e) * 40 + r] = vstage[i][e];
  }

  // ---- Q fragments live in VGPRs for the whole kernel ----
  const _Float16* qrow = Q + ((size_t)b * S1N + qbase + col) * DKN;
  v16h qfrag[4];
  #pragma unroll
  for (int kc = 0; kc < 4; ++kc) {
    V16H a;
    a.h[0] = *(const v8h*)(qrow + kc * 32 + 8 * h);
    a.h[1] = *(const v8h*)(qrow + kc * 32 + 16 + 8 * h);
    qfrag[kc] = a.v;
  }

  const float scale = 0.08838834764831845f;   // 1/sqrt(128)
  float m[8], l[8];
  v8f zero = {};
  v8f o[8];
  #pragma unroll
  for (int j = 0; j < 8; ++j) { m[j] = -1e30f; l[j] = 0.0f; }
  #pragma unroll
  for (int dc = 0; dc < 8; ++dc) o[dc] = zero;

  const int NT = S2N / KTILE;
  for (int it = 0; it < NT; ++it) {
    const int buf = it & 1;
#ifdef USE_ASYNC_COPY
    wait_async0();                          // my async K writes for buf landed
#endif
    __syncthreads();                        // all waves' tile writes visible

    // ---- issue prefetch of tile it+1 into buf^1 (overlaps with compute) ----
    if (it + 1 < NT) {
      const _Float16* kt1 = Kg + (size_t)(it + 1) * KTILE * DKN;
      const _Float16* vt1 = Vg + (size_t)(it + 1) * KTILE * DKN;
      #pragma unroll
      for (int i = 0; i < 4; ++i) {
        const int chunk = tid * 4 + i;
        const int r = chunk >> 4, c = chunk & 15;
#ifdef USE_ASYNC_COPY
        async_ld16(kt1 + (size_t)r * DKN + c * 8,
                   &Kt[buf ^ 1][r * 136 + c * 8]);
#else
        kstage[i] = *(const v8h*)(kt1 + (size_t)r * DKN + c * 8);
#endif
        vstage[i] = *(const v8h*)(vt1 + (size_t)r * DKN + c * 8);
      }
    }

    // ---- S = Q * K^T : batch-load 8 B-frags, then 8 WMMAs ----
    v16h bk[8];
    #pragma unroll
    for (int kc = 0; kc < 4; ++kc) {
      V16H b0, b1;
      const _Float16* p0 = &Kt[buf][col * 136 + kc * 32 + 8 * h];
      b0.h[0] = *(const v8h*)p0;  b0.h[1] = *(const v8h*)(p0 + 16);
      const _Float16* p1 = &Kt[buf][(16 + col) * 136 + kc * 32 + 8 * h];
      b1.h[0] = *(const v8h*)p1;  b1.h[1] = *(const v8h*)(p1 + 16);
      bk[2 * kc]     = b0.v;
      bk[2 * kc + 1] = b1.v;
    }
    v8f acc0 = zero, acc1 = zero;
    #pragma unroll
    for (int kc = 0; kc < 4; ++kc) {
      acc0 = wmma_f16(qfrag[kc], bk[2 * kc],     acc0);
      acc1 = wmma_f16(qfrag[kc], bk[2 * kc + 1], acc1);
    }

    // ---- online softmax (row reductions across the 16-lane C group) ----
    float p0v[8], p1v[8], alpha[8];
    #pragma unroll
    for (int j = 0; j < 8; ++j) {
      float v = fmaxf(acc0[j], acc1[j]) * scale;
      v = fmaxf(v, __shfl_xor(v, 1, 32));
      v = fmaxf(v, __shfl_xor(v, 2, 32));
      v = fmaxf(v, __shfl_xor(v, 4, 32));
      v = fmaxf(v, __shfl_xor(v, 8, 32));
      const float mn = fmaxf(m[j], v);
      p0v[j] = __expf(acc0[j] * scale - mn);
      p1v[j] = __expf(acc1[j] * scale - mn);
      float rs = p0v[j] + p1v[j];
      rs += __shfl_xor(rs, 1, 32);
      rs += __shfl_xor(rs, 2, 32);
      rs += __shfl_xor(rs, 4, 32);
      rs += __shfl_xor(rs, 8, 32);
      alpha[j] = __expf(m[j] - mn);
      l[j] = l[j] * alpha[j] + rs;
      m[j] = mn;
    }
    #pragma unroll
    for (int dc = 0; dc < 8; ++dc)
      #pragma unroll
      for (int j = 0; j < 8; ++j)
        o[dc][j] *= alpha[j];

    // ---- C-layout -> A-layout for P via wave-private LDS slab ----
    // (DS ops are in-order per wave; no block barrier needed here.)
    {
      _Float16* pw = &Pl[wave * 640];
      #pragma unroll
      for (int j = 0; j < 8; ++j) {
        _Float16* pr = pw + (j + 8 * h) * 40;
        pr[col]      = (_Float16)p0v[j];
        pr[16 + col] = (_Float16)p1v[j];
      }
    }
    V16H ap;
    {
      const _Float16* pp = &Pl[wave * 640 + col * 40];
      ap.h[0] = *(const v8h*)(pp + 8 * h);
      ap.h[1] = *(const v8h*)(pp + 16 + 8 * h);
    }

    // ---- O += P * V : batch-load 8 B-frags, then 8 WMMAs ----
    v16h bv[8];
    #pragma unroll
    for (int dc = 0; dc < 8; ++dc) {
      V16H bvr;
      const _Float16* vp = &Vt[buf][(dc * 16 + col) * 40 + 8 * h];
      bvr.h[0] = *(const v8h*)vp;  bvr.h[1] = *(const v8h*)(vp + 16);
      bv[dc] = bvr.v;
    }
    #pragma unroll
    for (int dc = 0; dc < 8; ++dc)
      o[dc] = wmma_f16(ap.v, bv[dc], o[dc]);

    // ---- drain staged prefetch into buf^1 (safe: readers passed barrier) ----
    if (it + 1 < NT) {
      #pragma unroll
      for (int i = 0; i < 4; ++i) {
        const int chunk = tid * 4 + i;
        const int r = chunk >> 4, c = chunk & 15;
#ifndef USE_ASYNC_COPY
        *(v8h*)&Kt[buf ^ 1][r * 136 + c * 8] = kstage[i];
#endif
        #pragma unroll
        for (int e = 0; e < 8; ++e)
          Vt[buf ^ 1][(c * 8 + e) * 40 + r] = vstage[i][e];
      }
    }
  }

  // ---- epilogue: Out[s][b][d] = O / l ----
  float inv[8];
  #pragma unroll
  for (int j = 0; j < 8; ++j) inv[j] = 1.0f / l[j];
  #pragma unroll
  for (int dc = 0; dc < 8; ++dc) {
    #pragma unroll
    for (int j = 0; j < 8; ++j) {
      const int srow = qbase + j + 8 * h;
      Out[((size_t)srow * BATCH + b) * DKN + dc * 16 + col] = o[dc][j] * inv[j];
    }
  }
}

// ---------------------------------------------------------------------------
extern "C" void kernel_launch(void* const* d_in, const int* in_sizes, int n_in,
                              void* d_out, int out_size, void* d_ws, size_t ws_size,
                              hipStream_t stream) {
  const float* x1 = (const float*)d_in[0];   // [S1, B, 100]
  const float* x2 = (const float*)d_in[1];   // [S2, B, 100]
  const float* Wq = (const float*)d_in[2];   // [100, 128]
  const float* Wk = (const float*)d_in[3];
  const float* Wv = (const float*)d_in[4];
  float* out = (float*)d_out;                // [S1, B, 128]

  // f16 workspace: Q | K | V, each B*S*128 halves (8 MB) -> 24 MB total
  _Float16* Qws = (_Float16*)d_ws;
  _Float16* Kws = Qws + (size_t)BATCH * S1N * DKN;
  _Float16* Vws = Kws + (size_t)BATCH * S2N * DKN;

  // 2048 row-tiles of 16 -> 512 blocks of 4 waves each
  qkv_proj_kernel<<<dim3(512), 128, 0, stream>>>(x1, Wq, Qws, S1N);
  qkv_proj_kernel<<<dim3(512), 128, 0, stream>>>(x2, Wk, Kws, S2N);
  qkv_proj_kernel<<<dim3(512), 128, 0, stream>>>(x2, Wv, Vws, S2N);

  flash_attn_kernel<<<dim3(S1N / 64, BATCH), 128, 0, stream>>>(Qws, Kws, Vws, out);
}
